// MHA_21689584845047
// MI455X (gfx1250) — matrix operand order repaired and checked by reading, not compiled, not run on hardware
//
#include <hip/hip_runtime.h>

// ---------------------------------------------------------------------------
// MHA forward for MI455X (gfx1250, wave32, WMMA).
// All GEMM work on v_wmma_f32_16x16x32_bf16 (f32 accumulate); attention K
// stream staged to LDS via gfx1250 async-load-to-LDS (ASYNCcnt) when the
// toolchain exposes it.
// ---------------------------------------------------------------------------

typedef __attribute__((ext_vector_type(16))) __bf16        v16bf;
typedef __attribute__((ext_vector_type(8)))  float         v8f;
typedef __attribute__((ext_vector_type(4)))  unsigned int  u32x4;
typedef __attribute__((ext_vector_type(4)))  int           v4i;

union Frag16 { v16bf v; unsigned int u[8]; };

static constexpr int Bsz   = 4;
static constexpr int Tseq  = 2048;
static constexpr int Cdim  = 1024;
static constexpr int Hn    = 16;
static constexpr int Dh    = 64;
static constexpr int Mrows = Bsz * Tseq;   // 8192
static constexpr int Nqkv  = 3 * Cdim;     // 3072

#define DEV __device__ __forceinline__

// ---- gfx1250 async global->LDS path (guarded; plain LDS copy fallback) ----
#if defined(__has_builtin)
#  if __has_builtin(__builtin_amdgcn_global_load_async_to_lds_b128)
#    define HAVE_ASYNC_LDS 1
#  endif
#endif
#ifndef HAVE_ASYNC_LDS
#  define HAVE_ASYNC_LDS 0
#endif

DEV void cp16_g2l(const unsigned short* g, unsigned short* l) {
#if HAVE_ASYNC_LDS
  __builtin_amdgcn_global_load_async_to_lds_b128(
      (__attribute__((address_space(1))) v4i*)g,
      (__attribute__((address_space(3))) v4i*)l, 0, 0);
#else
  *(u32x4*)l = *(const u32x4*)g;
#endif
}

DEV void async_wait0() {
#if HAVE_ASYNC_LDS
#  if defined(__has_builtin)
#    if __has_builtin(__builtin_amdgcn_s_wait_asynccnt)
  __builtin_amdgcn_s_wait_asynccnt(0);
#    else
  asm volatile("s_wait_asynccnt 0x0" ::: "memory");
#    endif
#  else
  asm volatile("s_wait_asynccnt 0x0" ::: "memory");
#  endif
#endif
}

DEV unsigned short f2bf(float f) {
  union { float f; unsigned int u; } v; v.f = f;
  unsigned int r = v.u + 0x7FFFu + ((v.u >> 16) & 1u);  // round-nearest-even
  return (unsigned short)(r >> 16);
}

// ---------------------------------------------------------------------------
// Prep kernels (one-shot, negligible vs 137 GFLOP of GEMM work)
// ---------------------------------------------------------------------------
__global__ void cast_f32_bf16(const float* __restrict__ in,
                              unsigned short* __restrict__ out, int n) {
  int i = blockIdx.x * blockDim.x + threadIdx.x;
  if (i < n) out[i] = f2bf(in[i]);
}

// w[k][n] (f32, K x N) -> wt[n][k] (bf16, N x K): GEMM B-loads become K-contiguous
__global__ void transpose_f32_bf16(const float* __restrict__ w,
                                   unsigned short* __restrict__ wt,
                                   int K, int N) {
  int i = blockIdx.x * blockDim.x + threadIdx.x;
  if (i < K * N) {
    int k = i / N, n = i - k * N;
    wt[(size_t)n * K + k] = f2bf(w[i]);
  }
}

// ---------------------------------------------------------------------------
// One wave computes a 16x64 output strip (4 tiles), A fragment reused 4x.
// A: row-major MxK bf16.  BT: row-major NxK bf16.
// ---------------------------------------------------------------------------
DEV void wmma_row16x64(const unsigned short* __restrict__ A,
                       const unsigned short* __restrict__ BT,
                       int lda, int ldb, int K, int m0, int n0, v8f acc[4]) {
  const int lane = threadIdx.x & 31;
  const int nl = lane & 15, hl = lane >> 4;
  const unsigned short* arow = A + (size_t)(m0 + nl) * lda;
  for (int kk = 0; kk < K; kk += 32) {
    Frag16 a;
#pragma unroll
    for (int p = 0; p < 8; ++p) {
      const int ko = hl * 16 + 2 * p;
      a.u[p] = *(const unsigned int*)(arow + kk + ko);
    }
#pragma unroll
    for (int t = 0; t < 4; ++t) {
      Frag16 b;
      const unsigned short* brow = BT + (size_t)(n0 + t * 16 + nl) * ldb + kk + hl * 16;
#pragma unroll
      for (int p = 0; p < 8; ++p) b.u[p] = *(const unsigned int*)(brow + 2 * p);
      acc[t] = __builtin_amdgcn_wmma_f32_16x16x32_bf16(
          false, a.v, false, b.v, (short)0, acc[t], false, false);
    }
  }
}

// ---------------------------------------------------------------------------
// QKV GEMM: qkv = x @ Wqkv + b, scattered into attention-friendly layouts:
//   Q,K -> [B,H,T,D]  (serves as B^T for S = Q K^T)
//   V   -> [B,H,D,T]  (serves as B^T for O = P V)
// ---------------------------------------------------------------------------
__global__ __launch_bounds__(128) void qkv_gemm(
    const unsigned short* __restrict__ Xb, const unsigned short* __restrict__ WT,
    const float* __restrict__ bias,
    unsigned short* __restrict__ qb, unsigned short* __restrict__ kb,
    unsigned short* __restrict__ vb) {
  const int wave = threadIdx.x >> 5;
  const int lane = threadIdx.x & 31;
  const int nl = lane & 15, hl = lane >> 4;
  const int m0 = blockIdx.x * 16;
  const int n0 = blockIdx.y * 256 + wave * 64;

  const v8f vzero = {0.f, 0.f, 0.f, 0.f, 0.f, 0.f, 0.f, 0.f};
  v8f acc[4];
#pragma unroll
  for (int t = 0; t < 4; ++t) acc[t] = vzero;

  wmma_row16x64(Xb, WT, Cdim, Cdim, Cdim, m0, n0, acc);

#pragma unroll
  for (int t = 0; t < 4; ++t) {
    const int col = n0 + t * 16 + nl;           // 64-col strips never straddle q/k/v or heads
    const int which = col >> 10, cc = col & 1023;
    const int hh = cc >> 6, dd = cc & 63;
    const float bv = bias[col];
#pragma unroll
    for (int r = 0; r < 8; ++r) {
      const int row = m0 + hl * 8 + r;
      const int bb = row >> 11, tq = row & 2047;
      const unsigned short val = f2bf(acc[t][r] + bv);
      if (which == 0)
        qb[((size_t)(bb * Hn + hh) * Tseq + tq) * Dh + dd] = val;
      else if (which == 1)
        kb[((size_t)(bb * Hn + hh) * Tseq + tq) * Dh + dd] = val;
      else
        vb[((size_t)(bb * Hn + hh) * Dh + dd) * Tseq + tq] = val;
    }
  }
}

// ---------------------------------------------------------------------------
// Flash attention: one wave owns a 16-row q tile, streams keys 32 at a time.
// Per chunk: 4 WMMA for S, register-level shuffle softmax, 4 WMMA for PV.
// K tiles are double-buffered in LDS via async global->LDS DMA (ASYNCcnt),
// overlapping the next chunk's fetch with the current chunk's WMMAs.
// ---------------------------------------------------------------------------
__global__ __launch_bounds__(128) void flash_attn(
    const unsigned short* __restrict__ Q, const unsigned short* __restrict__ Kt,
    const unsigned short* __restrict__ Vt, unsigned short* __restrict__ Y) {
  __shared__ __align__(16) unsigned short Klds[4][2][32][64];  // 32 KB
  __shared__ __align__(16) unsigned short Pbuf[4][16][32];     //  4 KB

  const int wave = threadIdx.x >> 5;
  const int lane = threadIdx.x & 31;
  const int nl = lane & 15, hl = lane >> 4;
  const int bh = blockIdx.y;
  const int q0 = blockIdx.x * 64 + wave * 16;

  const unsigned short* Qb = Q  + (size_t)bh * Tseq * Dh;
  const unsigned short* Kb = Kt + (size_t)bh * Tseq * Dh;
  const unsigned short* Vb = Vt + (size_t)bh * Dh * Tseq;

  // Q A-fragments for both D-halves, register-resident all loop
  Frag16 aq0, aq1;
#pragma unroll
  for (int p = 0; p < 8; ++p) {
    const int ko = hl * 16 + 2 * p;
    aq0.u[p] = *(const unsigned int*)(Qb + (size_t)(q0 + nl) * Dh + ko);
    aq1.u[p] = *(const unsigned int*)(Qb + (size_t)(q0 + nl) * Dh + 32 + ko);
  }

  const v8f vzero = {0.f, 0.f, 0.f, 0.f, 0.f, 0.f, 0.f, 0.f};
  v8f o[4];
#pragma unroll
  for (int t = 0; t < 4; ++t) o[t] = vzero;
  float m_i[8], l_i[8];
#pragma unroll
  for (int r = 0; r < 8; ++r) { m_i[r] = -3.0e38f; l_i[r] = 0.f; }

  // stage one 32x64 K tile (4 KB) into Klds[wave][buf]; 8 x 512B wave-copies
  auto stage = [&](int buf, int j0) {
    const int rK = lane >> 3, cK = (lane & 7) * 8;
    const unsigned short* gk = Kb + (size_t)(j0 + rK) * Dh + cK;
    unsigned short* lk = &Klds[wave][buf][rK][cK];
#pragma unroll
    for (int i = 0; i < 8; ++i) { cp16_g2l(gk, lk); gk += 4 * Dh; lk += 4 * 64; }
  };

  const int nchunks = (q0 + 47) >> 5;  // causal: keys up to q0+15
  stage(0, 0);
  for (int jc = 0; jc < nchunks; ++jc) {
    const int j0 = jc << 5;
    const int buf = jc & 1;
    async_wait0();                       // chunk jc resident in Klds[wave][buf]
    if (jc + 1 < nchunks) stage(buf ^ 1, j0 + 32);   // overlap next fetch

    // S = Q K^T for keys [j0, j0+15] and [j0+16, j0+31]  (K frags from LDS)
    v8f s0 = vzero, s1 = vzero;
    {
      Frag16 b;
#pragma unroll
      for (int p = 0; p < 8; ++p)
        b.u[p] = *(const unsigned int*)(&Klds[wave][buf][nl][hl * 16] + 2 * p);
      s0 = __builtin_amdgcn_wmma_f32_16x16x32_bf16(false, aq0.v, false, b.v, (short)0, s0, false, false);
#pragma unroll
      for (int p = 0; p < 8; ++p)
        b.u[p] = *(const unsigned int*)(&Klds[wave][buf][nl][32 + hl * 16] + 2 * p);
      s0 = __builtin_amdgcn_wmma_f32_16x16x32_bf16(false, aq1.v, false, b.v, (short)0, s0, false, false);
#pragma unroll
      for (int p = 0; p < 8; ++p)
        b.u[p] = *(const unsigned int*)(&Klds[wave][buf][16 + nl][hl * 16] + 2 * p);
      s1 = __builtin_amdgcn_wmma_f32_16x16x32_bf16(false, aq0.v, false, b.v, (short)0, s1, false, false);
#pragma unroll
      for (int p = 0; p < 8; ++p)
        b.u[p] = *(const unsigned int*)(&Klds[wave][buf][16 + nl][32 + hl * 16] + 2 * p);
      s1 = __builtin_amdgcn_wmma_f32_16x16x32_bf16(false, aq1.v, false, b.v, (short)0, s1, false, false);
    }

    // Register-level online softmax.  Row m = hl*8+r lives entirely in one
    // 16-lane half; xor-shuffles 1/2/4/8 reduce within that half.
    float crr[8];
#pragma unroll
    for (int r = 0; r < 8; ++r) {
      const int m = hl * 8 + r;
      const int qi = q0 + m;
      float x0 = s0[r] * 0.125f;  if (j0 + nl      > qi) x0 = -3.0e38f;
      float x1 = s1[r] * 0.125f;  if (j0 + 16 + nl > qi) x1 = -3.0e38f;

      float mx = fmaxf(x0, x1);
      mx = fmaxf(mx, __shfl_xor(mx, 1, 32));
      mx = fmaxf(mx, __shfl_xor(mx, 2, 32));
      mx = fmaxf(mx, __shfl_xor(mx, 4, 32));
      mx = fmaxf(mx, __shfl_xor(mx, 8, 32));

      const float nm = fmaxf(m_i[r], mx);
      const float corr = __expf(m_i[r] - nm);
      const float p0 = __expf(x0 - nm);
      const float p1 = __expf(x1 - nm);
      float sm = p0 + p1;
      sm += __shfl_xor(sm, 1, 32);
      sm += __shfl_xor(sm, 2, 32);
      sm += __shfl_xor(sm, 4, 32);
      sm += __shfl_xor(sm, 8, 32);

      l_i[r] = l_i[r] * corr + sm;
      m_i[r] = nm;
      crr[r] = corr;
      Pbuf[wave][m][nl]      = f2bf(p0);   // C-layout -> LDS
      Pbuf[wave][m][nl + 16] = f2bf(p1);
    }

#pragma unroll
    for (int t = 0; t < 4; ++t)
#pragma unroll
      for (int r = 0; r < 8; ++r) o[t][r] *= crr[r];

    // P back out of LDS in A-layout (16x32, K = 32 keys)
    Frag16 ap;
#pragma unroll
    for (int p = 0; p < 8; ++p) {
      const int ko = hl * 16 + 2 * p;
      ap.u[p] = *(const unsigned int*)&Pbuf[wave][nl][ko];
    }

    // O += P @ V   (V^T row-major [D][T]: K-contiguous, L2-resident)
#pragma unroll
    for (int t = 0; t < 4; ++t) {
      Frag16 bv;
#pragma unroll
      for (int p = 0; p < 8; ++p)
        bv.u[p] = *(const unsigned int*)(Vb + (size_t)(t * 16 + nl) * Tseq + j0 + hl * 16 + 2 * p);
      o[t] = __builtin_amdgcn_wmma_f32_16x16x32_bf16(false, ap.v, false, bv.v, (short)0, o[t], false, false);
    }
  }

  // normalize and write Y as bf16 in [B,T,C] row-major for the proj GEMM
  const int b = bh >> 4, h = bh & 15;
#pragma unroll
  for (int t = 0; t < 4; ++t) {
#pragma unroll
    for (int r = 0; r < 8; ++r) {
      const float linv = 1.0f / l_i[r];
      const int row = (b << 11) + q0 + hl * 8 + r;  // b*T + qi
      const int col = (h << 6) + t * 16 + nl;       // h*D + d
      Y[(size_t)row * Cdim + col] = f2bf(o[t][r] * linv);
    }
  }
}

// ---------------------------------------------------------------------------
// Proj GEMM: out = Y @ Wproj + b  (f32 output to d_out)
// ---------------------------------------------------------------------------
__global__ __launch_bounds__(128) void proj_gemm(
    const unsigned short* __restrict__ Yb, const unsigned short* __restrict__ WT,
    const float* __restrict__ bias, float* __restrict__ out) {
  const int wave = threadIdx.x >> 5;
  const int lane = threadIdx.x & 31;
  const int nl = lane & 15, hl = lane >> 4;
  const int m0 = blockIdx.x * 16;
  const int n0 = blockIdx.y * 256 + wave * 64;

  const v8f vzero = {0.f, 0.f, 0.f, 0.f, 0.f, 0.f, 0.f, 0.f};
  v8f acc[4];
#pragma unroll
  for (int t = 0; t < 4; ++t) acc[t] = vzero;

  wmma_row16x64(Yb, WT, Cdim, Cdim, Cdim, m0, n0, acc);

#pragma unroll
  for (int t = 0; t < 4; ++t) {
    const int col = n0 + t * 16 + nl;
    const float bv = bias[col];
#pragma unroll
    for (int r = 0; r < 8; ++r) {
      const int row = m0 + hl * 8 + r;
      out[(size_t)row * Cdim + col] = acc[t][r] + bv;
    }
  }
}

// ---------------------------------------------------------------------------
extern "C" void kernel_launch(void* const* d_in, const int* in_sizes, int n_in,
                              void* d_out, int out_size, void* d_ws, size_t ws_size,
                              hipStream_t stream) {
  (void)in_sizes; (void)n_in; (void)out_size; (void)ws_size;
  const float* x     = (const float*)d_in[0];
  const float* wqkv  = (const float*)d_in[1];
  const float* bqkv  = (const float*)d_in[2];
  const float* wproj = (const float*)d_in[3];
  const float* bproj = (const float*)d_in[4];
  float* out = (float*)d_out;

  char* ws = (char*)d_ws;
  size_t off = 0;
  auto take = [&](size_t elems) -> unsigned short* {
    unsigned short* p = (unsigned short*)(ws + off);
    off += (elems * sizeof(unsigned short) + 255) & ~(size_t)255;
    return p;
  };
  unsigned short* xb     = take((size_t)Mrows * Cdim);         // 16.8 MB
  unsigned short* wqkvT  = take((size_t)Nqkv * Cdim);          //  6.3 MB
  unsigned short* wprojT = take((size_t)Cdim * Cdim);          //  2.1 MB
  unsigned short* qb     = take((size_t)Bsz * Hn * Tseq * Dh); // 16.8 MB
  unsigned short* kb     = take((size_t)Bsz * Hn * Tseq * Dh); // 16.8 MB
  unsigned short* vb     = take((size_t)Bsz * Hn * Dh * Tseq); // 16.8 MB
  unsigned short* yb     = xb;  // x dead after qkv_gemm; reuse (stream-ordered)

  {
    const int n = Mrows * Cdim;
    cast_f32_bf16<<<(n + 255) / 256, 256, 0, stream>>>(x, xb, n);
  }
  {
    const int n = Cdim * Nqkv;
    transpose_f32_bf16<<<(n + 255) / 256, 256, 0, stream>>>(wqkv, wqkvT, Cdim, Nqkv);
  }
  {
    const int n = Cdim * Cdim;
    transpose_f32_bf16<<<(n + 255) / 256, 256, 0, stream>>>(wproj, wprojT, Cdim, Cdim);
  }

  qkv_gemm<<<dim3(Mrows / 16, Nqkv / 256), 128, 0, stream>>>(xb, wqkvT, bqkv, qb, kb, vb);
  flash_attn<<<dim3(Tseq / 64, Bsz * Hn), 128, 0, stream>>>(qb, kb, vb, yb);
  proj_gemm<<<dim3(Mrows / 16, Cdim / 256), 128, 0, stream>>>(yb, wprojT, bproj, out);
}
